// Decoder_70523363000803
// MI455X (gfx1250) — matrix operand order repaired and checked by reading, not compiled, not run on hardware
//
#include <hip/hip_runtime.h>
#include <hip/hip_bf16.h>
#include <math.h>

// ---------------------------------------------------------------------------
// Types for WMMA fragments (gfx1250, wave32)
// ---------------------------------------------------------------------------
typedef __attribute__((ext_vector_type(16))) _Float16 v16h;
typedef __attribute__((ext_vector_type(8)))  _Float16 v8h;
typedef __attribute__((ext_vector_type(8)))  float    v8f;

__device__ __forceinline__ float sigm(float x) { return 1.0f / (1.0f + __expf(-x)); }

// ---------------------------------------------------------------------------
// Fragment set for a 32x(16*NT) register tile, one K-block (K=32) deep.
// ---------------------------------------------------------------------------
template <int NT>
struct Frag {
  v8h aL[2], aH[2];
  v8h bL[NT], bH[NT];
};

template <int NT>
__device__ __forceinline__ void load_frag(Frag<NT>& f,
                                          const _Float16* const (&Ar)[2],
                                          const _Float16* const (&Wr)[NT],
                                          int kb) {
#pragma unroll
  for (int s = 0; s < 2; ++s) {
    f.aL[s] = *(const v8h*)(Ar[s] + kb);
    f.aH[s] = *(const v8h*)(Ar[s] + kb + 16);
  }
#pragma unroll
  for (int t = 0; t < NT; ++t) {
    f.bL[t] = *(const v8h*)(Wr[t] + kb);
    f.bH[t] = *(const v8h*)(Wr[t] + kb + 16);
  }
}

template <int NT>
__device__ __forceinline__ void mma_frag(v8f (&acc)[2][NT], const Frag<NT>& f) {
#pragma unroll
  for (int s = 0; s < 2; ++s) {
    v16h a;
#pragma unroll
    for (int e = 0; e < 8; ++e) { a[e] = f.aL[s][e]; a[8 + e] = f.aH[s][e]; }
#pragma unroll
    for (int t = 0; t < NT; ++t) {
      v16h b;
#pragma unroll
      for (int e = 0; e < 8; ++e) { b[e] = f.bL[t][e]; b[8 + e] = f.bH[t][e]; }
      acc[s][t] = __builtin_amdgcn_wmma_f32_16x16x32_f16(
          false, a, false, b, (short)0, acc[s][t], false, false);
    }
  }
}

// ---------------------------------------------------------------------------
// WMMA GEMM:  C[M,N] = act( A[M,K](f16) @ W[N,K]^T(f16) + bias[N] (+ C if acc) )
// Block = 128 threads = 4 waves.  Register tile per wave: 32 rows x 16*NT cols.
// grid = (N/(16*NT), M/128).  K % 32 == 0 and (K == 32 or K % 64 == 0);
// M % 128 == 0.  Ping-pong double buffering, K-loop unrolled by 2 so no
// register copies are needed and VMEM overlaps the matrix pipe.
// act: 0 = none, 1 = tanh.  Optional f16 mirror output Ch.
// ---------------------------------------------------------------------------
template <int NT>
__global__ __launch_bounds__(128) void wmma_gemm(
    const _Float16* __restrict__ A, const _Float16* __restrict__ W,
    const float* __restrict__ bias, float* __restrict__ C,
    _Float16* __restrict__ Ch, int M, int N, int K, int act, int acc_flag) {
  const int wave = threadIdx.x >> 5;
  const int lane = threadIdx.x & 31;
  const int lmod = lane & 15;
  const int lhi  = lane >> 4;
  const int m0 = blockIdx.y * 128 + wave * 32;   // wave owns rows m0 .. m0+31
  const int n0 = blockIdx.x * (16 * NT);

  const _Float16* Ar[2];
  Ar[0] = A + (size_t)(m0 + lmod) * K + lhi * 8;
  Ar[1] = A + (size_t)(m0 + 16 + lmod) * K + lhi * 8;
  const _Float16* Wr[NT];
#pragma unroll
  for (int t = 0; t < NT; ++t)
    Wr[t] = W + (size_t)(n0 + t * 16 + lmod) * K + lhi * 8;

  v8f zero = {};
  v8f acc[2][NT];
#pragma unroll
  for (int s = 0; s < 2; ++s)
#pragma unroll
    for (int t = 0; t < NT; ++t) acc[s][t] = zero;

  Frag<NT> f0, f1;
  load_frag<NT>(f0, Ar, Wr, 0);

  int kb = 0;
  for (; kb + 64 < K; kb += 64) {     // steady state: no branches, no copies
    load_frag<NT>(f1, Ar, Wr, kb + 32);
    mma_frag<NT>(acc, f0);
    load_frag<NT>(f0, Ar, Wr, kb + 64);
    mma_frag<NT>(acc, f1);
  }
  if (K - kb >= 64) {                 // tail: two K-blocks left
    load_frag<NT>(f1, Ar, Wr, kb + 32);
    mma_frag<NT>(acc, f0);
    mma_frag<NT>(acc, f1);
  } else {                            // tail: one K-block left (K == 32)
    mma_frag<NT>(acc, f0);
  }

#pragma unroll
  for (int s = 0; s < 2; ++s)
#pragma unroll
    for (int t = 0; t < NT; ++t) {
      const int n = n0 + t * 16 + lmod;
      const float bv = bias ? bias[n] : 0.0f;
#pragma unroll
      for (int r = 0; r < 8; ++r) {
        const int m = m0 + s * 16 + lhi * 8 + r;
        const size_t off = (size_t)m * N + n;
        float v = acc[s][t][r] + bv;
        if (acc_flag && C) v += C[off];
        if (act == 1) v = tanhf(v);
        if (C)  C[off]  = v;
        if (Ch) Ch[off] = (_Float16)v;
      }
    }
}

// ---------------------------------------------------------------------------
// Fused conv3x3(SAME) + ReLU + maxpool2x2.  One thread per output element.
// in: [B,IC,IH,IH]  w: [OC,IC,3,3]  out: [B,OC,IH/2,IH/2]
// ---------------------------------------------------------------------------
__global__ void conv_pool(const float* __restrict__ in_,
                          const float* __restrict__ w,
                          const float* __restrict__ bias,
                          float* __restrict__ out,
                          int IC, int OC, int IH, int total) {
  int idx = blockIdx.x * blockDim.x + threadIdx.x;
  if (idx >= total) return;
  const int OH = IH >> 1;
  int ox = idx % OH; int t = idx / OH;
  int oy = t % OH;   t /= OH;
  int o  = t % OC;   int b = t / OC;
  const float* inb = in_ + (size_t)b * IC * IH * IH;
  const float* wo  = w + (size_t)o * IC * 9;
  float mx = -1e30f;
  for (int py = 0; py < 2; ++py)
    for (int px = 0; px < 2; ++px) {
      const int cy = oy * 2 + py, cx = ox * 2 + px;
      float s = bias[o];
      for (int ic = 0; ic < IC; ++ic) {
        const float* ip = inb + (size_t)ic * IH * IH;
        const float* wi = wo + ic * 9;
        for (int ky = 0; ky < 3; ++ky) {
          const int iy = cy + ky - 1;
          if ((unsigned)iy >= (unsigned)IH) continue;
          for (int kx = 0; kx < 3; ++kx) {
            const int ix = cx + kx - 1;
            if ((unsigned)ix >= (unsigned)IH) continue;
            s += ip[iy * IH + ix] * wi[ky * 3 + kx];
          }
        }
      }
      s = fmaxf(s, 0.0f);
      mx = fmaxf(mx, s);
    }
  out[idx] = mx;
}

// f32 -> f16 convert with zero padding.  dst [Rd,Cd], src [Rs,Cs].
__global__ void cvt_pad_f16(const float* __restrict__ src, _Float16* __restrict__ dst,
                            int Rd, int Cd, int Rs, int Cs) {
  int idx = blockIdx.x * blockDim.x + threadIdx.x;
  int total = Rd * Cd;
  if (idx >= total) return;
  int c = idx % Cd, r = idx / Cd;
  float v = (r < Rs && c < Cs) ? src[(size_t)r * Cs + c] : 0.0f;
  dst[idx] = (_Float16)v;
}

// bsum = bih + bhh ; outbp = pad(out_b, 16)
__global__ void prep_bias(const float* bih, const float* bhh, const float* outb,
                          float* bsum, float* outbp) {
  int idx = blockIdx.x * blockDim.x + threadIdx.x;
  if (idx < 2048) bsum[idx] = bih[idx] + bhh[idx];
  else if (idx < 2064) { int k = idx - 2048; outbp[k] = (k < 11) ? outb[k] : 0.0f; }
}

// avg over channels + img_parts (f16 transpose).  feats [B,32,64].
// thread per (b, j):  avg16[b,j] = mean_c feats[b,c,j];  ip16[b,j,c] = feats[b,c,j]
__global__ void avg_parts(const float* __restrict__ feats,
                          _Float16* __restrict__ avg16,
                          _Float16* __restrict__ ip16) {
  int idx = blockIdx.x * blockDim.x + threadIdx.x;
  if (idx >= 1024 * 64) return;
  int j = idx & 63, b = idx >> 6;
  const float* fb = feats + (size_t)b * 2048;
  _Float16* row = ip16 + (size_t)idx * 32;
  float s = 0.0f;
  for (int c = 0; c < 32; ++c) {
    float v = fb[c * 64 + j];
    s += v;
    row[c] = (_Float16)v;
  }
  avg16[idx] = (_Float16)(s * (1.0f / 32.0f));
}

// score[b,j] = v_b + sum_d tanh(Ws[b,j,d] + Uh[b,d]) * v[d]
__global__ void attn_score(const float* __restrict__ Ws, const float* __restrict__ Uh,
                           const float* __restrict__ vw, const float* __restrict__ vb,
                           float* __restrict__ score) {
  int idx = blockIdx.x * blockDim.x + threadIdx.x;
  if (idx >= 1024 * 64) return;
  int b = idx >> 6;
  const float* wsr = Ws + (size_t)idx * 128;
  const float* uh  = Uh + (size_t)b * 128;
  float s = vb[0];
  for (int d = 0; d < 128; ++d) s += tanhf(wsr[d] + uh[d]) * vw[d];
  score[idx] = s;
}

// softmax over 64 + context: block of 64 threads per batch element
__global__ void softmax_context(const float* __restrict__ score,
                                const float* __restrict__ feats,
                                float* __restrict__ context) {
  __shared__ float salpha[64];
  __shared__ float stmp[64];
  const int b = blockIdx.x, j = threadIdx.x;
  float s = score[b * 64 + j];
  stmp[j] = s; __syncthreads();
  for (int off = 32; off > 0; off >>= 1) {
    if (j < off) stmp[j] = fmaxf(stmp[j], stmp[j + off]);
    __syncthreads();
  }
  const float mx = stmp[0]; __syncthreads();
  float e = __expf(s - mx);
  salpha[j] = e; stmp[j] = e; __syncthreads();
  for (int off = 32; off > 0; off >>= 1) {
    if (j < off) stmp[j] += stmp[j + off];
    __syncthreads();
  }
  const float inv = 1.0f / stmp[0]; __syncthreads();
  if (j < 32) {
    const float* fb = feats + (size_t)b * 2048 + j * 64;  // channel j row over hw
    float acc = 0.0f;
    for (int n = 0; n < 64; ++n) acc += salpha[n] * fb[n];
    context[b * 32 + j] = acc * inv;
  }
}

// inp16[b, 0:12] = label[b, t+1, :]; [12:44] = context*sigmoid(betapre); [44:64]=0
__global__ void build_inp(const float* __restrict__ label,
                          const float* __restrict__ context,
                          const float* __restrict__ betapre,
                          _Float16* __restrict__ inp16, int t) {
  int idx = blockIdx.x * blockDim.x + threadIdx.x;
  if (idx >= 1024 * 64) return;
  int k = idx & 63, b = idx >> 6;
  float v = 0.0f;
  if (k < 12) v = label[(size_t)b * 16 * 12 + (size_t)(t + 1) * 12 + k];
  else if (k < 44) {
    int c = k - 12;
    v = context[b * 32 + c] * sigm(betapre[b * 32 + c]);
  }
  inp16[idx] = (_Float16)v;
}

// LSTM cell: gates [B,2048] (i,f,g,o); updates c (f32) and h (f16)
__global__ void lstm_cell(const float* __restrict__ gates,
                          float* __restrict__ c, _Float16* __restrict__ h16) {
  int idx = blockIdx.x * blockDim.x + threadIdx.x;
  if (idx >= 1024 * 512) return;
  int u = idx & 511, b = idx >> 9;
  const float* g = gates + (size_t)b * 2048;
  float i = sigm(g[u]);
  float f = sigm(g[512 + u]);
  float gg = tanhf(g[1024 + u]);
  float o = sigm(g[1536 + u]);
  float cc = f * c[idx] + i * gg;
  c[idx] = cc;
  h16[idx] = (_Float16)(o * tanhf(cc));
}

// log_softmax over first 11 of outtmp[b,16] -> d_out[b, t, 0:11]
__global__ void logsoftmax_out(const float* __restrict__ outtmp,
                               float* __restrict__ dout, int t) {
  int b = blockIdx.x * blockDim.x + threadIdx.x;
  if (b >= 1024) return;
  const float* r = outtmp + (size_t)b * 16;
  float mx = r[0];
  for (int k = 1; k < 11; ++k) mx = fmaxf(mx, r[k]);
  float s = 0.0f;
  for (int k = 0; k < 11; ++k) s += __expf(r[k] - mx);
  const float lse = mx + logf(s);
  float* ob = dout + (size_t)b * 15 * 11 + (size_t)t * 11;
  for (int k = 0; k < 11; ++k) ob[k] = r[k] - lse;
}

// ---------------------------------------------------------------------------
extern "C" void kernel_launch(void* const* d_in, const int* in_sizes, int n_in,
                              void* d_out, int out_size, void* d_ws, size_t ws_size,
                              hipStream_t stream) {
  (void)in_sizes; (void)n_in; (void)out_size; (void)ws_size;
  const float* img     = (const float*)d_in[0];
  const float* label   = (const float*)d_in[1];
  const float* conv1_w = (const float*)d_in[2];
  const float* conv1_b = (const float*)d_in[3];
  const float* conv2_w = (const float*)d_in[4];
  const float* conv2_b = (const float*)d_in[5];
  const float* conv3_w = (const float*)d_in[6];
  const float* conv3_b = (const float*)d_in[7];
  const float* W_w     = (const float*)d_in[8];
  const float* W_b     = (const float*)d_in[9];
  const float* U_w     = (const float*)d_in[10];
  const float* U_b     = (const float*)d_in[11];
  const float* v_w     = (const float*)d_in[12];
  const float* v_b     = (const float*)d_in[13];
  const float* beta_w  = (const float*)d_in[14];
  const float* beta_b  = (const float*)d_in[15];
  const float* wih     = (const float*)d_in[16];
  const float* whh     = (const float*)d_in[17];
  const float* bih     = (const float*)d_in[18];
  const float* bhh     = (const float*)d_in[19];
  const float* ih_w    = (const float*)d_in[20];
  const float* ih_b    = (const float*)d_in[21];
  const float* ic_w    = (const float*)d_in[22];
  const float* ic_b    = (const float*)d_in[23];
  const float* out_w   = (const float*)d_in[24];
  const float* out_b   = (const float*)d_in[25];
  float* dout = (float*)d_out;

  // ---- workspace carving (256B aligned) ----
  size_t off = 0;
  auto alloc = [&](size_t bytes) -> void* {
    off = (off + 255) & ~(size_t)255;
    void* p = (char*)d_ws + off;
    off += bytes;
    return p;
  };
  float* pooled1 = (float*)alloc((size_t)1024 * 8 * 32 * 32 * 4);   // 33.5MB; reused as Ws
  float* pooled2 = (float*)alloc((size_t)1024 * 16 * 16 * 16 * 4);  // 16.8MB; reused below
  float* feats   = (float*)alloc((size_t)1024 * 32 * 64 * 4);       // 8.4MB
  _Float16* avg16  = (_Float16*)alloc((size_t)1024 * 64 * 2);
  _Float16* ip16   = (_Float16*)alloc((size_t)1024 * 64 * 32 * 2);
  _Float16* Ww16   = (_Float16*)alloc((size_t)128 * 32 * 2);
  _Float16* U16    = (_Float16*)alloc((size_t)128 * 512 * 2);
  _Float16* beta16 = (_Float16*)alloc((size_t)32 * 512 * 2);
  _Float16* whh16  = (_Float16*)alloc((size_t)2048 * 512 * 2);
  _Float16* wih16  = (_Float16*)alloc((size_t)2048 * 64 * 2);  // K padded 44->64
  _Float16* ihw16  = (_Float16*)alloc((size_t)512 * 64 * 2);
  _Float16* icw16  = (_Float16*)alloc((size_t)512 * 64 * 2);
  _Float16* outw16 = (_Float16*)alloc((size_t)16 * 512 * 2);   // N padded 11->16
  float* bsum   = (float*)alloc(2048 * 4);
  float* outbp  = (float*)alloc(16 * 4);
  float* Uh     = (float*)alloc((size_t)1024 * 128 * 4);
  float* score  = (float*)alloc((size_t)1024 * 64 * 4);
  float* context= (float*)alloc((size_t)1024 * 32 * 4);
  float* betapre= (float*)alloc((size_t)1024 * 32 * 4);
  _Float16* inp16 = (_Float16*)alloc((size_t)1024 * 64 * 2);
  float* outtmp = (float*)alloc((size_t)1024 * 16 * 4);
  // aliases (pooled1/pooled2 free after encoder finishes)
  float* Ws     = pooled1;                                   // [B,64,128] f32 (33.5MB)
  float* gates  = pooled2;                                   // [B,2048] f32 (8MB)
  float* cstate = pooled2 + (size_t)2 * 1024 * 1024;         // [B,512] f32 (2MB)
  _Float16* h16 = (_Float16*)(pooled2 + (size_t)2 * 1024 * 1024 + 512 * 1024); // 1MB

  const int TPB = 256;
  auto blocks = [](int n) { return (n + 255) / 256; };

  // ---- weight conversion to f16 (every call; deterministic) ----
  cvt_pad_f16<<<blocks(128 * 32), TPB, 0, stream>>>(W_w, Ww16, 128, 32, 128, 32);
  cvt_pad_f16<<<blocks(128 * 512), TPB, 0, stream>>>(U_w, U16, 128, 512, 128, 512);
  cvt_pad_f16<<<blocks(32 * 512), TPB, 0, stream>>>(beta_w, beta16, 32, 512, 32, 512);
  cvt_pad_f16<<<blocks(2048 * 512), TPB, 0, stream>>>(whh, whh16, 2048, 512, 2048, 512);
  cvt_pad_f16<<<blocks(2048 * 64), TPB, 0, stream>>>(wih, wih16, 2048, 64, 2048, 44);
  cvt_pad_f16<<<blocks(512 * 64), TPB, 0, stream>>>(ih_w, ihw16, 512, 64, 512, 64);
  cvt_pad_f16<<<blocks(512 * 64), TPB, 0, stream>>>(ic_w, icw16, 512, 64, 512, 64);
  cvt_pad_f16<<<blocks(16 * 512), TPB, 0, stream>>>(out_w, outw16, 16, 512, 11, 512);
  prep_bias<<<blocks(2064), TPB, 0, stream>>>(bih, bhh, out_b, bsum, outbp);

  // ---- encoder: fused conv+relu+pool x3 ----
  {
    int t1 = 1024 * 8 * 32 * 32;
    conv_pool<<<blocks(t1), TPB, 0, stream>>>(img, conv1_w, conv1_b, pooled1, 1, 8, 64, t1);
    int t2 = 1024 * 16 * 16 * 16;
    conv_pool<<<blocks(t2), TPB, 0, stream>>>(pooled1, conv2_w, conv2_b, pooled2, 8, 16, 32, t2);
    int t3 = 1024 * 32 * 8 * 8;
    conv_pool<<<blocks(t3), TPB, 0, stream>>>(pooled2, conv3_w, conv3_b, feats, 16, 32, 16, t3);
  }
  avg_parts<<<blocks(1024 * 64), TPB, 0, stream>>>(feats, avg16, ip16);

  // Ws = img_parts @ W_w^T + W_b : M=65536, N=128, K=32 (overwrites pooled1)
  wmma_gemm<4><<<dim3(2, 512), 128, 0, stream>>>(ip16, Ww16, W_b, Ws, nullptr,
                                                 65536, 128, 32, 0, 0);
  // h0 = tanh(avg @ ih_w^T + b) -> f16 only ; c0 = tanh(avg @ ic_w^T + b) -> f32
  wmma_gemm<4><<<dim3(8, 8), 128, 0, stream>>>(avg16, ihw16, ih_b, nullptr, h16,
                                               1024, 512, 64, 1, 0);
  wmma_gemm<4><<<dim3(8, 8), 128, 0, stream>>>(avg16, icw16, ic_b, cstate, nullptr,
                                               1024, 512, 64, 1, 0);

  // ---- decoder: 15 teacher-forced steps ----
  for (int t = 0; t < 15; ++t) {
    // Uh = h @ U^T + U_b : [1024,128], K=512
    wmma_gemm<4><<<dim3(2, 8), 128, 0, stream>>>(h16, U16, U_b, Uh, nullptr,
                                                 1024, 128, 512, 0, 0);
    attn_score<<<blocks(1024 * 64), TPB, 0, stream>>>(Ws, Uh, v_w, v_b, score);
    softmax_context<<<1024, 64, 0, stream>>>(score, feats, context);
    // beta_pre = h @ beta_w^T + beta_b : [1024,32]
    wmma_gemm<2><<<dim3(1, 8), 128, 0, stream>>>(h16, beta16, beta_b, betapre, nullptr,
                                                 1024, 32, 512, 0, 0);
    build_inp<<<blocks(1024 * 64), TPB, 0, stream>>>(label, context, betapre, inp16, t);
    // gates = inp @ wih^T + (bih+bhh) : [1024,2048], K=64
    wmma_gemm<4><<<dim3(32, 8), 128, 0, stream>>>(inp16, wih16, bsum, gates, nullptr,
                                                  1024, 2048, 64, 0, 0);
    // gates += h @ whh^T : K=512 (the dominant GEMM)
    wmma_gemm<4><<<dim3(32, 8), 128, 0, stream>>>(h16, whh16, nullptr, gates, nullptr,
                                                  1024, 2048, 512, 0, 1);
    lstm_cell<<<blocks(1024 * 512), TPB, 0, stream>>>(gates, cstate, h16);
    // out = h @ out_w^T + out_b : [1024,16(pad)], K=512
    wmma_gemm<1><<<dim3(1, 8), 128, 0, stream>>>(h16, outw16, outbp, outtmp, nullptr,
                                                 1024, 16, 512, 0, 0);
    logsoftmax_out<<<blocks(1024), TPB, 0, stream>>>(outtmp, dout, t);
  }
}
